// DynNSAttention_42116449304973
// MI455X (gfx1250) — matrix operand order repaired and checked by reading, not compiled, not run on hardware
//
#include <hip/hip_runtime.h>
#include <stdint.h>

// Problem constants (B=1)
#define TT    2048
#define CC    1024
#define HH    16
#define HSZ   64
#define NBLK  32
#define BLKSZ 64
#define SSEL  8
#define WINSZ 128
#define SCALE_F 0.125f   // 1/sqrt(64)
#define NEG_BIG (-1e30f)

typedef __bf16 bf16;
typedef __attribute__((ext_vector_type(8)))  __bf16 v8bf;
typedef __attribute__((ext_vector_type(16))) __bf16 v16bf;
typedef __attribute__((ext_vector_type(8)))  float  v8f;
typedef __attribute__((ext_vector_type(4)))  unsigned int u32x4;
typedef __attribute__((ext_vector_type(8)))  int i32x8;
typedef __attribute__((ext_vector_type(4)))  int i32x4;

#define WMMA_BF16(a, b, c) \
  __builtin_amdgcn_wmma_f32_16x16x32_bf16(false, (a), false, (b), (short)0, (c), false, false)

static __device__ __forceinline__ v8bf ld8(const bf16* p) {
  return *reinterpret_cast<const v8bf*>(p);
}
static __device__ __forceinline__ v16bf mk16(v8bf a, v8bf b) {
  return __builtin_shufflevector(a, b, 0,1,2,3,4,5,6,7,8,9,10,11,12,13,14,15);
}

// LDS byte offset of a __shared__ object (generic addr low 32 bits map to LDS)
static __device__ __forceinline__ unsigned lds_off(const void* p) {
  return (unsigned)(uintptr_t)p;
}

// Tensor Data Mover: 2D bf16 tile (tileD1 rows x tileD0 elems, row stride strideD0
// elements) from global -> packed LDS tile.  D# layout per CDNA5 ISA 08_async_tensor
// (group0: count/lds_addr/global_addr/type=2 ; group1: data_size=2B, dims, strides).
static __device__ __forceinline__ void tdm_load_2d(unsigned ldsAddr, const void* gaddr,
                                                   unsigned tensorD0, unsigned tensorD1,
                                                   unsigned long long strideD0,
                                                   unsigned tileD0, unsigned tileD1) {
  unsigned long long ga = (unsigned long long)(uintptr_t)gaddr;
  u32x4 g0;
  g0[0] = 1u;                                       // count=1, user descriptor
  g0[1] = ldsAddr;                                  // lds_addr
  g0[2] = (unsigned)(ga & 0xFFFFFFFFu);             // global_addr[31:0]
  g0[3] = (unsigned)((ga >> 32) & 0x1FFFFFFu) | (2u << 30);  // addr[56:32] | type=2
  i32x8 g1;
  g1[0] = (int)(1u << 16);                                          // data_size=1 (2B)
  g1[1] = (int)((tensorD0 & 0xFFFFu) << 16);                        // tensor_dim0 lo
  g1[2] = (int)(((tensorD0 >> 16) & 0xFFFFu) | ((tensorD1 & 0xFFFFu) << 16));
  g1[3] = (int)(((tensorD1 >> 16) & 0xFFFFu) | ((tileD0 & 0xFFFFu) << 16));
  g1[4] = (int)(tileD1 & 0xFFFFu);                                  // tile_dim1, dim2=0
  g1[5] = (int)(strideD0 & 0xFFFFFFFFull);                          // dim0_stride lo32
  g1[6] = (int)((strideD0 >> 32) & 0xFFFFull);                      // hi16 ; dim1_stride=0
  g1[7] = 0;
  // 6-arg toolchain variant: (g0, g1, g2, g3, g4, cpol)
  __builtin_amdgcn_tensor_load_to_lds(g0, g1, (i32x4)0, (i32x4)0, (i32x8)0, 0);
}

// ---------------------------------------------------------------- conversions
__global__ void cvt_bf16_kernel(const float* __restrict__ in, bf16* __restrict__ out, int n) {
  int i = blockIdx.x * blockDim.x + threadIdx.x;
  if (i < n) out[i] = (bf16)in[i];
}

// in: [K][N] f32  ->  out: [N][K] bf16
__global__ void transpose_cvt_kernel(const float* __restrict__ in, bf16* __restrict__ out,
                                     int K, int N) {
  int i = blockIdx.x * blockDim.x + threadIdx.x;
  if (i >= K * N) return;
  int k = i / N, n = i % N;
  out[(size_t)n * K + k] = (bf16)in[(size_t)k * N + n];
}

// ---------------------------------------------------------------- WMMA GEMM
// Y[M x N] = A[M x K](bf16) * Bt[N x K](bf16)^T + bias   (register double-buffered)
// MODE 0: store bf16 head-major [H][M][64]   (col = h*64+d)
// MODE 1: store bf16 transposed [col][M]     (i.e. [H][HS][T])
// MODE 2: store f32 [M][N]
template <int NSUB, int MODE>
__global__ __launch_bounds__(128)
void gemm_bf16_kernel(const bf16* __restrict__ A, const bf16* __restrict__ Bt,
                      const float* __restrict__ bias, void* __restrict__ outp,
                      int M, int N, int K) {
  const int wave  = threadIdx.x >> 5;
  const int lane  = threadIdx.x & 31;
  const int lhalf = lane & 15;
  const int hi    = (lane >> 4) & 1;
  const int m0    = blockIdx.x * 64 + wave * 16;
  const int n0    = blockIdx.y * (16 * NSUB);
  const int arow  = m0 + lhalf;

  const bf16* apBase = A + (size_t)arow * K + (hi ? 8 : 0);
  auto loadA = [&](int k0) { return mk16(ld8(apBase + k0), ld8(apBase + k0 + 16)); };
  auto loadB = [&](int nt, int k0) {
    const int col = n0 + nt * 16 + lhalf;
    const bf16* bp = Bt + (size_t)col * K + k0 + (hi ? 8 : 0);
    return mk16(ld8(bp), ld8(bp + 16));
  };

  v8f acc[NSUB] = {};
  v16bf aCur = loadA(0);
  v16bf bCur[NSUB];
  #pragma unroll
  for (int nt = 0; nt < NSUB; ++nt) bCur[nt] = loadB(nt, 0);

  for (int k0 = 0; k0 < K; k0 += 32) {
    v16bf aNxt = aCur;
    v16bf bNxt[NSUB];
    #pragma unroll
    for (int nt = 0; nt < NSUB; ++nt) bNxt[nt] = bCur[nt];
    if (k0 + 32 < K) {
      aNxt = loadA(k0 + 32);
      #pragma unroll
      for (int nt = 0; nt < NSUB; ++nt) bNxt[nt] = loadB(nt, k0 + 32);
    }
    #pragma unroll
    for (int nt = 0; nt < NSUB; ++nt) acc[nt] = WMMA_BF16(aCur, bCur[nt], acc[nt]);
    aCur = aNxt;
    #pragma unroll
    for (int nt = 0; nt < NSUB; ++nt) bCur[nt] = bNxt[nt];
  }

  #pragma unroll
  for (int nt = 0; nt < NSUB; ++nt) {
    const int col = n0 + nt * 16 + lhalf;
    const float bv = bias ? bias[col] : 0.0f;
    #pragma unroll
    for (int r = 0; r < 8; ++r) {
      const int row = m0 + r + (hi ? 8 : 0);
      const float v = acc[nt][r] + bv;
      if (MODE == 2) {
        ((float*)outp)[(size_t)row * N + col] = v;
      } else if (MODE == 0) {
        const int hh = col >> 6, d = col & 63;
        ((bf16*)outp)[(((size_t)hh * M) + row) * HSZ + d] = (bf16)v;
      } else {
        ((bf16*)outp)[(size_t)col * M + row] = (bf16)v;
      }
    }
  }
}

// ---------------------------------------------------------------- gates (N=3)
__global__ void gates_kernel(const float* __restrict__ X, const float* __restrict__ Wg,
                             const float* __restrict__ bg, float* __restrict__ gates) {
  int t = blockIdx.x * blockDim.x + threadIdx.x;
  if (t >= TT) return;
  float a0 = bg[0], a1 = bg[1], a2 = bg[2];
  const float* x = X + (size_t)t * CC;
  for (int c = 0; c < CC; ++c) {
    float xv = x[c];
    a0 += xv * Wg[c * 3 + 0];
    a1 += xv * Wg[c * 3 + 1];
    a2 += xv * Wg[c * 3 + 2];
  }
  float m = fmaxf(a0, fmaxf(a1, a2));
  float e0 = __expf(a0 - m), e1 = __expf(a1 - m), e2 = __expf(a2 - m);
  float z = e0 + e1 + e2;
  gates[t * 3 + 0] = e0 / z;
  gates[t * 3 + 1] = e1 / z;
  gates[t * 3 + 2] = e2 / z;
}

// ---------------------------------------------------------------- block means
__global__ void comp_means_kernel(const bf16* __restrict__ Kb, const bf16* __restrict__ Vt,
                                  bf16* __restrict__ kcomp, bf16* __restrict__ vcompT) {
  int i = blockIdx.x * blockDim.x + threadIdx.x;
  if (i >= HH * NBLK * HSZ) return;
  int d = i & 63, n = (i >> 6) & 31, h = i >> 11;
  float sk = 0.f, sv = 0.f;
  for (int j = 0; j < BLKSZ; ++j) {
    sk += (float)Kb[(((size_t)h * TT) + n * BLKSZ + j) * HSZ + d];
    sv += (float)Vt[(((size_t)h * HSZ) + d) * TT + n * BLKSZ + j];
  }
  kcomp[(((size_t)h * NBLK) + n) * HSZ + d]  = (bf16)(sk * (1.0f / BLKSZ));
  vcompT[(((size_t)h * HSZ) + d) * NBLK + n] = (bf16)(sv * (1.0f / BLKSZ));
}

// P(16 x 32*nFrag, LDS row-major stride 72) * V-cols -> accumulate into O[4] (hs tiles)
static __device__ __forceinline__ void pv_accum(v8f O[4], const bf16* P,
                                                const bf16* Vcols, int keyBase,
                                                int keyStride, int nFrag,
                                                int hi, int lhalf) {
  #pragma unroll
  for (int f = 0; f < 2; ++f) {
    if (f >= nFrag) break;
    const bf16* pr = P + (size_t)lhalf * 72 + f * 32 + (hi ? 8 : 0);
    v16bf pa = mk16(ld8(pr), ld8(pr + 16));
    #pragma unroll
    for (int nt = 0; nt < 4; ++nt) {
      const bf16* vp = Vcols + (size_t)(nt * 16 + lhalf) * keyStride
                     + keyBase + f * 32 + (hi ? 8 : 0);
      v16bf vb = mk16(ld8(vp), ld8(vp + 16));
      O[nt] = WMMA_BF16(pa, vb, O[nt]);
    }
  }
}

// score tile helper: Q(16x64) x Ktile-rows(16x64)^T from LDS tile (row stride 64)
static __device__ __forceinline__ v8f score_tile(v16bf aQ0, v16bf aQ1,
                                                 const bf16* krow, int hi) {
  v16bf b0 = mk16(ld8(krow + (hi ? 8 : 0)),      ld8(krow + (hi ? 8 : 0) + 16));
  v16bf b1 = mk16(ld8(krow + 32 + (hi ? 8 : 0)), ld8(krow + 32 + (hi ? 8 : 0) + 16));
  v8f acc = {};
  acc = WMMA_BF16(aQ0, b0, acc);
  acc = WMMA_BF16(aQ1, b1, acc);
  return acc;
}

// ---------------------------------------------------------------- attention
__global__ __launch_bounds__(128)
void attn_kernel(const bf16* __restrict__ Qb, const bf16* __restrict__ Kb,
                 const bf16* __restrict__ Vt, const bf16* __restrict__ kcomp,
                 const bf16* __restrict__ vcompT, const float* __restrict__ gates,
                 const int* __restrict__ topIdx,
                 bf16* __restrict__ attnOut, float* __restrict__ entOut) {
  __shared__ bf16 Plds[4][16][72];        // per-wave P transpose staging
  __shared__ bf16 Ktile[BLKSZ][HSZ];      // TDM-staged K block (rows = keys)
  __shared__ bf16 Vtile[HSZ][BLKSZ];      // TDM-staged V block (rows = hs, cols = keys)

  const int h     = blockIdx.y;
  const int wave  = threadIdx.x >> 5;
  const int lane  = threadIdx.x & 31;
  const int lhalf = lane & 15;
  const int hi    = (lane >> 4) & 1;
  const int qt0   = blockIdx.x * 64;
  const int qt    = qt0 + wave * 16;

  const bf16* Qh = Qb + (size_t)h * TT * HSZ;
  const bf16* Kh = Kb + (size_t)h * TT * HSZ;
  const bf16* Vh = Vt + (size_t)h * HSZ * TT;
  bf16* Pme = &Plds[wave][0][0];

  // Q A-fragments (hs 0..31 and 32..63)
  const bf16* qp = Qh + (size_t)(qt + lhalf) * HSZ;
  v16bf aQ0 = mk16(ld8(qp + (hi ? 8 : 0)),      ld8(qp + (hi ? 8 : 0) + 16));
  v16bf aQ1 = mk16(ld8(qp + 32 + (hi ? 8 : 0)), ld8(qp + 32 + (hi ? 8 : 0) + 16));

  v8f Oslide[4] = {}, Osel[4] = {}, Ocomp[4] = {};
  float mSl[8], zSl[8], mSe[8], zSe[8];
  #pragma unroll
  for (int r = 0; r < 8; ++r) { mSl[r] = mSe[r] = NEG_BIG; zSl[r] = zSe[r] = 0.f; }

  const int nlo  = (qt > WINSZ) ? (qt - WINSZ) / BLKSZ : 0;   // per-wave slide range
  const int nhi  = (qt + 15) >> 6;
  const int unlo = (qt0 > WINSZ) ? (qt0 - WINSZ) / BLKSZ : 0; // workgroup union range
  const int unhi = (qt0 + 63) >> 6;

  // -------- phase A: full-score pass -> entropies + sliding-window attention
  #pragma unroll 1
  for (int n = 0; n < NBLK; ++n) {
    const bool needV = (n >= unlo) && (n <= unhi);
    __syncthreads();                                   // prior tile consumers done
    if (wave == 0) {
      tdm_load_2d(lds_off(&Ktile[0][0]), Kh + (size_t)n * BLKSZ * HSZ,
                  HSZ, BLKSZ, HSZ, HSZ, BLKSZ);
      if (needV)
        tdm_load_2d(lds_off(&Vtile[0][0]), Vh + (size_t)n * BLKSZ,
                    BLKSZ, HSZ, TT, BLKSZ, HSZ);
      __builtin_amdgcn_s_wait_tensorcnt(0);
    }
    __syncthreads();                                   // tiles visible to all waves

    v8f s[4];
    #pragma unroll
    for (int t4 = 0; t4 < 4; ++t4) {
      v8f acc = score_tile(aQ0, aQ1, &Ktile[t4 * 16 + lhalf][0], hi);
      #pragma unroll
      for (int r = 0; r < 8; ++r) acc[r] *= SCALE_F;
      s[t4] = acc;
    }

    // entropy over the 64 keys of this block (per query row)
    #pragma unroll
    for (int r = 0; r < 8; ++r) {
      float lm = fmaxf(fmaxf(s[0][r], s[1][r]), fmaxf(s[2][r], s[3][r]));
      #pragma unroll
      for (int o = 1; o < 16; o <<= 1) lm = fmaxf(lm, __shfl_xor(lm, o, 32));
      float e = 0.f, w = 0.f;
      #pragma unroll
      for (int t4 = 0; t4 < 4; ++t4) {
        float d = s[t4][r] - lm; float ex = __expf(d);
        e += ex; w += d * ex;
      }
      #pragma unroll
      for (int o = 1; o < 16; o <<= 1) { e += __shfl_xor(e, o, 32); w += __shfl_xor(w, o, 32); }
      if (lhalf == 0) {
        int row = qt + r + (hi ? 8 : 0);
        entOut[(((size_t)h * TT) + row) * NBLK + n] = __logf(e) - w / e;
      }
    }

    // sliding-window online softmax (per-wave relevant blocks only)
    if (n >= nlo && n <= nhi) {
      float fr[8];
      #pragma unroll
      for (int r = 0; r < 8; ++r) {
        const int ti = qt + r + (hi ? 8 : 0);
        #pragma unroll
        for (int t4 = 0; t4 < 4; ++t4) {
          int j = n * BLKSZ + t4 * 16 + lhalf;
          bool ok = (j <= ti) && (j + WINSZ >= ti);
          s[t4][r] = ok ? s[t4][r] : NEG_BIG;
        }
        float lm = fmaxf(fmaxf(s[0][r], s[1][r]), fmaxf(s[2][r], s[3][r]));
        #pragma unroll
        for (int o = 1; o < 16; o <<= 1) lm = fmaxf(lm, __shfl_xor(lm, o, 32));
        float nm = fmaxf(mSl[r], lm);
        float f  = __expf(mSl[r] - nm);
        mSl[r] = nm;
        float ps = 0.f;
        #pragma unroll
        for (int t4 = 0; t4 < 4; ++t4) { float p = __expf(s[t4][r] - nm); ps += p; s[t4][r] = p; }
        #pragma unroll
        for (int o = 1; o < 16; o <<= 1) ps += __shfl_xor(ps, o, 32);
        zSl[r] = zSl[r] * f + ps;
        fr[r] = f;
      }
      #pragma unroll
      for (int nt = 0; nt < 4; ++nt)
        #pragma unroll
        for (int r = 0; r < 8; ++r) Oslide[nt][r] *= fr[r];
      #pragma unroll
      for (int t4 = 0; t4 < 4; ++t4)
        #pragma unroll
        for (int r = 0; r < 8; ++r)
          Plds[wave][r + (hi ? 8 : 0)][t4 * 16 + lhalf] = (bf16)s[t4][r];
      asm volatile("s_wait_dscnt 0" ::: "memory");
      pv_accum(Oslide, Pme, &Vtile[0][0], 0, BLKSZ, 2, hi, lhalf);
    }
  }

  // -------- phase B: selected blocks (mask is on concatenated index)
  #pragma unroll 1
  for (int si = 0; si < SSEL; ++si) {
    const int blk = topIdx[h * SSEL + si];
    __syncthreads();
    if (wave == 0) {
      tdm_load_2d(lds_off(&Ktile[0][0]), Kh + (size_t)blk * BLKSZ * HSZ,
                  HSZ, BLKSZ, HSZ, HSZ, BLKSZ);
      tdm_load_2d(lds_off(&Vtile[0][0]), Vh + (size_t)blk * BLKSZ,
                  BLKSZ, HSZ, TT, BLKSZ, HSZ);
      __builtin_amdgcn_s_wait_tensorcnt(0);
    }
    __syncthreads();

    v8f s[4];
    #pragma unroll
    for (int t4 = 0; t4 < 4; ++t4) {
      v8f acc = score_tile(aQ0, aQ1, &Ktile[t4 * 16 + lhalf][0], hi);
      #pragma unroll
      for (int r = 0; r < 8; ++r) acc[r] *= SCALE_F;
      s[t4] = acc;
    }
    float fr[8];
    #pragma unroll
    for (int r = 0; r < 8; ++r) {
      const int ti = qt + r + (hi ? 8 : 0);
      #pragma unroll
      for (int t4 = 0; t4 < 4; ++t4) {
        int jc = si * BLKSZ + t4 * 16 + lhalf;
        s[t4][r] = (jc <= ti) ? s[t4][r] : NEG_BIG;
      }
      float lm = fmaxf(fmaxf(s[0][r], s[1][r]), fmaxf(s[2][r], s[3][r]));
      #pragma unroll
      for (int o = 1; o < 16; o <<= 1) lm = fmaxf(lm, __shfl_xor(lm, o, 32));
      float nm = fmaxf(mSe[r], lm);
      float f  = __expf(mSe[r] - nm);
      mSe[r] = nm;
      float ps = 0.f;
      #pragma unroll
      for (int t4 = 0; t4 < 4; ++t4) { float p = __expf(s[t4][r] - nm); ps += p; s[t4][r] = p; }
      #pragma unroll
      for (int o = 1; o < 16; o <<= 1) ps += __shfl_xor(ps, o, 32);
      zSe[r] = zSe[r] * f + ps;
      fr[r] = f;
    }
    #pragma unroll
    for (int nt = 0; nt < 4; ++nt)
      #pragma unroll
      for (int r = 0; r < 8; ++r) Osel[nt][r] *= fr[r];
    #pragma unroll
    for (int t4 = 0; t4 < 4; ++t4)
      #pragma unroll
      for (int r = 0; r < 8; ++r)
        Plds[wave][r + (hi ? 8 : 0)][t4 * 16 + lhalf] = (bf16)s[t4][r];
    asm volatile("s_wait_dscnt 0" ::: "memory");
    pv_accum(Osel, Pme, &Vtile[0][0], 0, BLKSZ, 2, hi, lhalf);
  }

  // -------- phase C: compressed attention (32 block-mean keys, one shot)
  float zC[8];
  {
    v8f s2[2];
    #pragma unroll
    for (int t4 = 0; t4 < 2; ++t4) {
      const bf16* kr = kcomp + (((size_t)h * NBLK) + t4 * 16 + lhalf) * HSZ;
      v8f acc = score_tile(aQ0, aQ1, kr, hi);
      #pragma unroll
      for (int r = 0; r < 8; ++r) {
        const int ti = qt + r + (hi ? 8 : 0);
        const int kidx = t4 * 16 + lhalf;
        acc[r] = (kidx <= ti) ? acc[r] * SCALE_F : NEG_BIG;
      }
      s2[t4] = acc;
    }
    #pragma unroll
    for (int r = 0; r < 8; ++r) {
      float lm = fmaxf(s2[0][r], s2[1][r]);
      #pragma unroll
      for (int o = 1; o < 16; o <<= 1) lm = fmaxf(lm, __shfl_xor(lm, o, 32));
      float ps = 0.f;
      #pragma unroll
      for (int t4 = 0; t4 < 2; ++t4) { float p = __expf(s2[t4][r] - lm); ps += p; s2[t4][r] = p; }
      #pragma unroll
      for (int o = 1; o < 16; o <<= 1) ps += __shfl_xor(ps, o, 32);
      zC[r] = ps;
    }
    #pragma unroll
    for (int t4 = 0; t4 < 2; ++t4)
      #pragma unroll
      for (int r = 0; r < 8; ++r)
        Plds[wave][r + (hi ? 8 : 0)][t4 * 16 + lhalf] = (bf16)s2[t4][r];
    asm volatile("s_wait_dscnt 0" ::: "memory");
    pv_accum(Ocomp, Pme, vcompT + (size_t)h * HSZ * NBLK, 0, NBLK, 1, hi, lhalf);
  }

  // -------- merge with gates, store bf16 [T][C] for the output projection
  #pragma unroll
  for (int r = 0; r < 8; ++r) {
    const int row = qt + r + (hi ? 8 : 0);
    const float g0 = gates[row * 3 + 0];
    const float g1 = gates[row * 3 + 1];
    const float g2 = gates[row * 3 + 2];
    #pragma unroll
    for (int nt = 0; nt < 4; ++nt) {
      float v = g0 * (Oslide[nt][r] / zSl[r])
              + g1 * (Osel[nt][r]   / zSe[r])
              + g2 * (Ocomp[nt][r]  / zC[r]);
      attnOut[(size_t)row * CC + h * HSZ + nt * 16 + lhalf] = (bf16)v;
    }
  }
}

// ---------------------------------------------------------------- launch
extern "C" void kernel_launch(void* const* d_in, const int* in_sizes, int n_in,
                              void* d_out, int out_size, void* d_ws, size_t ws_size,
                              hipStream_t stream) {
  (void)in_sizes; (void)n_in; (void)out_size; (void)ws_size;

  const float* X   = (const float*)d_in[0];
  const int* topIdx = (const int*)d_in[1];
  const float* Wq = (const float*)d_in[2];   const float* bq = (const float*)d_in[3];
  const float* Wk = (const float*)d_in[4];   const float* bk = (const float*)d_in[5];
  const float* Wv = (const float*)d_in[6];   const float* bv = (const float*)d_in[7];
  const float* Wo = (const float*)d_in[8];   const float* bo = (const float*)d_in[9];
  const float* Wg = (const float*)d_in[10];  const float* bg = (const float*)d_in[11];
  const float* Wr = (const float*)d_in[12];  const float* br = (const float*)d_in[13];

  float* outY   = (float*)d_out;                 // [T][C]
  float* outR   = outY + (size_t)TT * CC;        // [T][NB]
  float* outEnt = outR + (size_t)TT * NBLK;      // [H][T][NB]

  char* wsp = (char*)d_ws;
  auto alloc = [&](size_t bytes) {
    char* p = wsp;
    wsp += (bytes + 255) & ~(size_t)255;
    return p;
  };
  bf16* Xb      = (bf16*)alloc((size_t)TT * CC * 2);
  bf16* WqT     = (bf16*)alloc((size_t)CC * CC * 2);
  bf16* WkT     = (bf16*)alloc((size_t)CC * CC * 2);
  bf16* WvT     = (bf16*)alloc((size_t)CC * CC * 2);
  bf16* WoT     = (bf16*)alloc((size_t)CC * CC * 2);
  bf16* WrT     = (bf16*)alloc((size_t)NBLK * CC * 2);
  bf16* Qb      = (bf16*)alloc((size_t)TT * CC * 2);   // [H][T][HS]
  bf16* Kb      = (bf16*)alloc((size_t)TT * CC * 2);   // [H][T][HS]
  bf16* Vt      = (bf16*)alloc((size_t)TT * CC * 2);   // [H][HS][T]
  bf16* kcomp   = (bf16*)alloc((size_t)HH * NBLK * HSZ * 2);
  bf16* vcompT  = (bf16*)alloc((size_t)HH * HSZ * NBLK * 2);
  float* gatesB = (float*)alloc((size_t)TT * 3 * 4);
  bf16* attnOut = (bf16*)alloc((size_t)TT * CC * 2);

  cvt_bf16_kernel<<<(TT * CC) / 256, 256, 0, stream>>>(X, Xb, TT * CC);
  transpose_cvt_kernel<<<(CC * CC) / 256, 256, 0, stream>>>(Wq, WqT, CC, CC);
  transpose_cvt_kernel<<<(CC * CC) / 256, 256, 0, stream>>>(Wk, WkT, CC, CC);
  transpose_cvt_kernel<<<(CC * CC) / 256, 256, 0, stream>>>(Wv, WvT, CC, CC);
  transpose_cvt_kernel<<<(CC * CC) / 256, 256, 0, stream>>>(Wo, WoT, CC, CC);
  transpose_cvt_kernel<<<(CC * NBLK) / 256, 256, 0, stream>>>(Wr, WrT, CC, NBLK);

  dim3 gg(TT / 64, CC / 64);
  gemm_bf16_kernel<4, 0><<<gg, 128, 0, stream>>>(Xb, WqT, bq, Qb, TT, CC, CC);
  gemm_bf16_kernel<4, 0><<<gg, 128, 0, stream>>>(Xb, WkT, bk, Kb, TT, CC, CC);
  gemm_bf16_kernel<4, 1><<<gg, 128, 0, stream>>>(Xb, WvT, bv, Vt, TT, CC, CC);
  gemm_bf16_kernel<2, 2><<<dim3(TT / 64, 1), 128, 0, stream>>>(Xb, WrT, br, outR, TT, NBLK, CC);

  gates_kernel<<<TT / 128, 128, 0, stream>>>(X, Wg, bg, gatesB);
  comp_means_kernel<<<(HH * NBLK * HSZ) / 256, 256, 0, stream>>>(Kb, Vt, kcomp, vcompT);

  attn_kernel<<<dim3(TT / 64, HH), 128, 0, stream>>>(Qb, Kb, Vt, kcomp, vcompT,
                                                     gatesB, topIdx, attnOut, outEnt);

  gemm_bf16_kernel<4, 2><<<gg, 128, 0, stream>>>(attnOut, WoT, bo, outY, TT, CC, CC);
}
